// ScatteredExperts_83803401879802
// MI455X (gfx1250) — matrix operand order repaired and checked by reading, not compiled, not run on hardware
//
#include <hip/hip_runtime.h>

// ---------------- problem constants (from reference) ----------------
#define NTOK 4096
#define DIN  1024
#define DOUT 1024
#define NEXP 8
#define TOPK 2
#define NPOS (NTOK * TOPK)     // 8192 sorted routed positions

// ---------------- tiling ----------------
#define TM 128                 // rows (sorted positions) per block
#define TN 128                 // output columns per block
#define TK 32                  // K step (bf16 WMMA depth)
#define NSTEP (DIN / TK)       // 32 K-steps
#define LDST 40                // halfs per LDS row (80 B: 16B-aligned, bank-spread)

typedef __attribute__((ext_vector_type(16))) __bf16 v16bf;
typedef __attribute__((ext_vector_type(8)))  float  v8f;
typedef int v4i __attribute__((vector_size(16)));    // matches builtin param type

union Frag16 {
    uint4 u[2];
    v16bf v;
};

// ---------------- async-to-LDS staging (CDNA5 path) ----------------
#if __has_builtin(__builtin_amdgcn_global_load_async_to_lds_b128)
#define ASYNC_VIA_BUILTIN 1
#else
#define ASYNC_VIA_BUILTIN 0
#endif

__device__ __forceinline__ void async_b128(const unsigned short* g, unsigned short* l) {
#if ASYNC_VIA_BUILTIN
    // signature (probe-derived): (global v4i* src, lds v4i* dst, int offset, int cpol)
    auto gp = (__attribute__((address_space(1))) v4i*)(size_t)g;
    auto lp = (__attribute__((address_space(3))) v4i*)(size_t)l;
    __builtin_amdgcn_global_load_async_to_lds_b128(gp, lp, 0, 0);
#else
    // GV mode: vdst = LDS byte address (low 32 bits of generic LDS addr), vaddr = 64-bit global
    unsigned lds = (unsigned)(size_t)l;
    asm volatile("global_load_async_to_lds_b128 %0, %1, off offset:0"
                 :: "v"(lds), "v"(g) : "memory");
#endif
}

__device__ __forceinline__ void wait_async0() {
#if __has_builtin(__builtin_amdgcn_s_wait_asynccnt)
    __builtin_amdgcn_s_wait_asynccnt(0);
#else
    asm volatile("s_wait_asynccnt 0x0" ::: "memory");
#endif
}

__device__ __forceinline__ unsigned short f2bf(float f) {
    unsigned int u = __float_as_uint(f);
    return (unsigned short)((u + (0x7FFFu + ((u >> 16) & 1u))) >> 16);  // RNE
}
__device__ __forceinline__ unsigned int pack_bf16x2(float a, float b) {
    unsigned int ua = __float_as_uint(a);
    unsigned int ub = __float_as_uint(b);
    ua = (ua + (0x7FFFu + ((ua >> 16) & 1u))) >> 16;
    ub = (ub + (0x7FFFu + ((ub >> 16) & 1u))) & 0xFFFF0000u;
    return ua | ub;
}

// ---------------- pre-pass: x f32 -> bf16 (same layout) ----------------
__global__ __launch_bounds__(256) void convert_x_bf16(const float* __restrict__ x,
                                                      unsigned int* __restrict__ xb32) {
    int i = blockIdx.x * 256 + threadIdx.x;           // 4 floats per thread
    float4 v = *(const float4*)(x + (size_t)i * 4);
    uint2 p;
    p.x = pack_bf16x2(v.x, v.y);
    p.y = pack_bf16x2(v.z, v.w);
    *(uint2*)(xb32 + (size_t)i * 2) = p;
}

// ---- pre-pass: weight f32 [E][DIN][DOUT] -> bf16 K-tiled transpose ----
// wt layout: [E][DIN/TK][DOUT][TK] halfs  (per K-step, a DOUT x TK slab is contiguous)
__global__ __launch_bounds__(256) void transpose_w_bf16(const float* __restrict__ w,
                                                        unsigned short* __restrict__ wt) {
    __shared__ __align__(16) unsigned short T[32][36];
    const int e  = blockIdx.z;
    const int kb = blockIdx.x;
    const int c0 = blockIdx.y * 32;
    const int tid = threadIdx.x;
    const int r = tid >> 3;             // 0..31
    const int g = tid & 7;              // 0..7

    float4 v = *(const float4*)(w + ((size_t)e * DIN + kb * 32 + r) * DOUT + c0 + g * 4);
    T[g * 4 + 0][r] = f2bf(v.x);
    T[g * 4 + 1][r] = f2bf(v.y);
    T[g * 4 + 2][r] = f2bf(v.z);
    T[g * 4 + 3][r] = f2bf(v.w);
    __syncthreads();

    uint2 p = *(const uint2*)&T[r][g * 4];
    *(uint2*)(wt + (((size_t)e * NSTEP + kb) * DOUT + (c0 + r)) * TK + g * 4) = p;
}

// ---------------- zero the output (harness poisons d_out) ----------------
__global__ __launch_bounds__(256) void moe_zero_out(float* __restrict__ out) {
    int i = (blockIdx.x * 256 + threadIdx.x) * 4;
    *(float4*)(out + i) = make_float4(0.f, 0.f, 0.f, 0.f);
}

// ---------------- one expert pass over the full K dimension ----------------
// Wave layout: mw = wave>>1 (4 M-groups of 32 rows), nw = wave&1 (2 N-groups of 64 cols).
// Each wave: 2 A-fragments x 4 B-fragments -> 8 WMMAs from 12 ds_load_b128.
// B (and A when unmasked) staged with GLOBAL_LOAD_ASYNC_TO_LDS_B128 (ASYNCcnt).
template<bool MASKED>
__device__ __forceinline__ void expert_pass(
        int e,
        const unsigned short* __restrict__ xb,
        const unsigned short* __restrict__ wt,
        int colBase,
        const int* s_tok, const int* s_bin,
        unsigned short (&As)[2][TM][LDST],
        unsigned short (&Bs)[2][TN][LDST],
        v8f (&acc)[2][4])
{
    const int tid  = threadIdx.x;
    const int wave = tid >> 5;
    const int lane = tid & 31;
    const int mw   = wave >> 1;
    const int nw   = wave & 1;

    // staging: thread covers 16 halfs (32 B) of each 8 KB tile
    const int ar = tid >> 1;
    const int ah = (tid & 1) * 16;
    const unsigned short* aPtr = xb + (size_t)s_tok[ar] * DIN + ah;
    const unsigned short* bPtr = wt + ((size_t)e * NSTEP * DOUT + colBase) * TK + tid * 16;

    unsigned int amask = 0xFFFFFFFFu;
    if (MASKED) amask = (s_bin[ar] == e) ? 0xFFFFFFFFu : 0u;

    // fragment addresses (constant across steps)
    const int aRow0 = mw * 32 + (lane & 15);
    const int kh    = (lane >> 4) * 8;
    const int kbb   = (lane >> 4) * 8;

    // ---- prologue: stage step 0 into buffer 0 ----
    uint4 a0, a1;
    if (MASKED) {
        a0 = *(const uint4*)(aPtr);
        a1 = *(const uint4*)(aPtr + 8);
        a0.x &= amask; a0.y &= amask; a0.z &= amask; a0.w &= amask;
        a1.x &= amask; a1.y &= amask; a1.z &= amask; a1.w &= amask;
    }
    __syncthreads();   // previous pass fully done with both buffers
    if (MASKED) {
        *(uint4*)&As[0][ar][ah]     = a0;
        *(uint4*)&As[0][ar][ah + 8] = a1;
    } else {
        async_b128(aPtr,     &As[0][ar][ah]);
        async_b128(aPtr + 8, &As[0][ar][ah + 8]);
    }
    async_b128(bPtr,     &Bs[0][ar][ah]);    // same (row,half) mapping: col = tid>>1
    async_b128(bPtr + 8, &Bs[0][ar][ah + 8]);

    for (int s = 0; s < NSTEP; ++s) {
        wait_async0();                 // our async writes (step s data) are in LDS
        __syncthreads();               // buf[s&1] ready; buf[(s+1)&1] free
        const int cur = s & 1;
        const int nxt = cur ^ 1;

        // issue staging for step s+1 (latency hidden under WMMAs)
        if (s + 1 < NSTEP) {
            if (MASKED) {
                a0 = *(const uint4*)(aPtr + (s + 1) * TK);
                a1 = *(const uint4*)(aPtr + (s + 1) * TK + 8);
                a0.x &= amask; a0.y &= amask; a0.z &= amask; a0.w &= amask;
                a1.x &= amask; a1.y &= amask; a1.z &= amask; a1.w &= amask;
            } else {
                async_b128(aPtr + (s + 1) * TK,     &As[nxt][ar][ah]);
                async_b128(aPtr + (s + 1) * TK + 8, &As[nxt][ar][ah + 8]);
            }
            async_b128(bPtr + (size_t)(s + 1) * DOUT * TK,     &Bs[nxt][ar][ah]);
            async_b128(bPtr + (size_t)(s + 1) * DOUT * TK + 8, &Bs[nxt][ar][ah + 8]);
        }

        // ---- compute step s from buf[cur]: 12 ds_load_b128 -> 8 WMMAs ----
        Frag16 af0, af1;
        af0.u[0] = *(const uint4*)&As[cur][aRow0][kh];
        af0.u[1] = *(const uint4*)&As[cur][aRow0][kh + 16];
        af1.u[0] = *(const uint4*)&As[cur][aRow0 + 16][kh];
        af1.u[1] = *(const uint4*)&As[cur][aRow0 + 16][kh + 16];
        #pragma unroll
        for (int n = 0; n < 4; ++n) {
            const int col = nw * 64 + n * 16 + (lane & 15);
            Frag16 b;
            b.u[0] = *(const uint4*)&Bs[cur][col][kbb];
            b.u[1] = *(const uint4*)&Bs[cur][col][kbb + 16];
            acc[0][n] = __builtin_amdgcn_wmma_f32_16x16x32_bf16(
                false, af0.v, false, b.v, (short)0, acc[0][n], false, false);
            acc[1][n] = __builtin_amdgcn_wmma_f32_16x16x32_bf16(
                false, af1.v, false, b.v, (short)0, acc[1][n], false, false);
        }

        // masked A path stages via registers -> LDS after compute
        if (MASKED && (s + 1 < NSTEP)) {
            *(uint4*)&As[nxt][ar][ah]     = a0;
            *(uint4*)&As[nxt][ar][ah + 8] = a1;
        }
    }
}

// ---------------- main grouped-GEMM kernel (bf16 WMMA, double-buffered) ----------------
__global__ __launch_bounds__(256) void moe_scattered_gemm(
        const unsigned short* __restrict__ xb,   // [NTOK][DIN] bf16
        const unsigned short* __restrict__ wt,   // [E][NSTEP][DOUT][TK] bf16
        const int*   __restrict__ bin_ids,       // [NPOS]
        const int*   __restrict__ indices,       // [NPOS]
        const float* __restrict__ gates,         // [NTOK][TOPK]
        float*       __restrict__ out)           // [NTOK][DOUT]
{
    __shared__ __align__(16) unsigned short As[2][TM][LDST];
    __shared__ __align__(16) unsigned short Bs[2][TN][LDST];
    __shared__ int   s_tok[TM];
    __shared__ int   s_bin[TM];
    __shared__ float s_gate[TM];

    const int tid     = threadIdx.x;
    const int wave    = tid >> 5;
    const int lane    = tid & 31;
    const int rowBase = blockIdx.y * TM;
    const int colBase = blockIdx.x * TN;

    if (tid < TM) {
        int p    = rowBase + tid;
        int idx  = indices[p];
        int tok  = idx >> 1;            // TOPK == 2
        int slot = idx & 1;
        s_tok[tid]  = tok;
        s_bin[tid]  = bin_ids[p];
        s_gate[tid] = gates[tok * TOPK + slot];
    }
    __syncthreads();

    const int eFirst = s_bin[0];
    const int eLast  = s_bin[TM - 1];

    v8f acc[2][4] = {};

    if (eFirst == eLast) {
        // common case: whole tile belongs to one expert -> no masking, no e-loop
        expert_pass<false>(eFirst, xb, wt, colBase, s_tok, s_bin, As, Bs, acc);
    } else {
        // boundary tile: masked accumulation per expert in [eFirst, eLast]
        for (int e = eFirst; e <= eLast; ++e)
            expert_pass<true>(e, xb, wt, colBase, s_tok, s_bin, As, Bs, acc);
    }

    // ---- epilogue: gate-scale + scatter-add (C layout: VGPR j -> row j + 8*(lane>>4)) ----
    const int mw = wave >> 1;
    const int nw = wave & 1;
    #pragma unroll
    for (int mi = 0; mi < 2; ++mi) {
        #pragma unroll
        for (int n = 0; n < 4; ++n) {
            const int col = colBase + nw * 64 + n * 16 + (lane & 15);
            #pragma unroll
            for (int j = 0; j < 8; ++j) {
                const int rLocal = mw * 32 + mi * 16 + j + ((lane >> 4) * 8);
                atomicAdd(&out[(size_t)s_tok[rLocal] * DOUT + col],
                          acc[mi][n][j] * s_gate[rLocal]);
            }
        }
    }
}

extern "C" void kernel_launch(void* const* d_in, const int* in_sizes, int n_in,
                              void* d_out, int out_size, void* d_ws, size_t ws_size,
                              hipStream_t stream) {
    const float* x        = (const float*)d_in[0];
    const float* w        = (const float*)d_in[1];
    const int*   bin_ids  = (const int*)  d_in[2];
    const int*   indices  = (const int*)  d_in[3];
    // d_in[4] padded_block_idxs, d_in[5] expert_offsets: unused
    const float* gates    = (const float*)d_in[6];
    float*       out      = (float*)d_out;

    // workspace layout: [0, 16MB) weight bf16 (K-tiled transpose); [16MB, 24MB) x bf16
    unsigned short* wt = (unsigned short*)d_ws;
    unsigned short* xb = wt + (size_t)NEXP * DIN * DOUT;

    convert_x_bf16<<<(NTOK * DIN) / 1024, 256, 0, stream>>>(x, (unsigned int*)xb);
    transpose_w_bf16<<<dim3(NSTEP, DOUT / 32, NEXP), 256, 0, stream>>>(w, wt);
    moe_zero_out<<<(NTOK * DOUT) / 1024, 256, 0, stream>>>(out);

    dim3 grid(DOUT / TN, NPOS / TM);   // (8, 64)
    moe_scattered_gemm<<<grid, 256, 0, stream>>>(xb, wt, bin_ids, indices, gates, out);
}
